// SpatialMambaBlock_46677704573690
// MI455X (gfx1250) — compile-verified, matrix operand order checked by reading, not run on hardware
//
#include <hip/hip_runtime.h>

// SpatialMambaBlock for MI455X (gfx1250, wave32).
// Token GEMMs: v_wmma_f32_16x16x32_f16, 2x2 register blocking, K-step 64
// (8 WMMAs per barrier pair). Selective scan: 16-way chunked 3-pass scan.
#define DIMC    192
#define DINC    384
#define DTRANK_ 12
#define NSTATE_ 16
#define BATCH_  2
#define HH      56
#define WW      56
#define LSEQ    (HH * WW)          // 3136
#define ROWS_   (BATCH_ * LSEQ)    // 6272 = 49 * 128
#define NCHUNK  16
#define CLEN    (LSEQ / NCHUNK)    // 196

typedef _Float16 h16v __attribute__((ext_vector_type(16)));
typedef _Float16 h8v  __attribute__((ext_vector_type(8)));
typedef float    f8v  __attribute__((ext_vector_type(8)));

__device__ __forceinline__ float siluf(float x)     { return x / (1.0f + __expf(-x)); }
__device__ __forceinline__ float softplusf_(float x){ return (x > 20.0f) ? x : log1pf(__expf(x)); }

// Fragment gather per ISA 16x32 f16 A/B layout:
// lane<16: row m=lane, K = {0..7, 16..23}; lane>=16: row m=lane-16, K = {8..15, 24..31}
__device__ __forceinline__ h16v load_frag(const _Float16* rowp, int half8)
{
    h8v lo = *(const h8v*)(rowp + half8);
    h8v hi = *(const h8v*)(rowp + half8 + 16);
    h16v f;
    #pragma unroll
    for (int i = 0; i < 8; ++i) { f[i] = lo[i]; f[i + 8] = hi[i]; }
    return f;
}

// ---------------------------------------------------------------------------
// Generic WMMA GEMM:  C[ROWS_, N] = A[ROWS_, K] * Wt[N, K]^T  (+ epilogue)
// Block: 256 threads = 8 waves; workgroup tile M=128, N=64, K-step 64;
// each wave computes a 32x32 sub-tile; 8 WMMAs per K iteration per wave.
// epi: 0 = none, 1 = bias + exact GELU, 2 = bias + residual add
// Requires K % 64 == 0 (K in {192, 384, 768} here).
// ---------------------------------------------------------------------------
__global__ __launch_bounds__(256)
void k_gemm_wmma(const float* __restrict__ A, const float* __restrict__ Wt,
                 float* __restrict__ C, const float* __restrict__ bias,
                 const float* __restrict__ resid,
                 int N, int K, int ldC, int epi)
{
    __shared__ _Float16 lA[128 * 72];  // 128 rows x 64 cols, padded stride 72
    __shared__ _Float16 lB[64 * 72];   //  64 rows x 64 cols, padded stride 72

    const int tid  = threadIdx.x;
    const int lane = tid & 31;
    const int wave = tid >> 5;
    const int wm   = wave >> 1;        // 0..3 -> rows wm*32
    const int wn   = wave & 1;         // 0..1 -> cols wn*32
    const int m0   = blockIdx.y * 128;
    const int n0   = blockIdx.x * 64;

    f8v acc00 = {}, acc01 = {}, acc10 = {}, acc11 = {};

    for (int k0 = 0; k0 < K; k0 += 64) {
        // Stage A tile 128x64 (f32 -> f16): 1024 8-elem tasks, 4 per thread
        #pragma unroll
        for (int i = 0; i < 4; ++i) {
            const int task = tid + 256 * i;
            const int row  = task >> 3;
            const int cg   = (task & 7) * 8;
            const float4* p = (const float4*)(A + (size_t)(m0 + row) * K + (k0 + cg));
            float4 v0 = p[0], v1 = p[1];
            _Float16* dst = &lA[row * 72 + cg];
            dst[0] = (_Float16)v0.x; dst[1] = (_Float16)v0.y;
            dst[2] = (_Float16)v0.z; dst[3] = (_Float16)v0.w;
            dst[4] = (_Float16)v1.x; dst[5] = (_Float16)v1.y;
            dst[6] = (_Float16)v1.z; dst[7] = (_Float16)v1.w;
        }
        // Stage B tile 64x64 (weights [N,K] row-major), zero-pad beyond N
        #pragma unroll
        for (int i = 0; i < 2; ++i) {
            const int task = tid + 256 * i;
            const int row  = task >> 3;
            const int cg   = (task & 7) * 8;
            _Float16* dst = &lB[row * 72 + cg];
            const int n = n0 + row;
            if (n < N) {
                const float4* p = (const float4*)(Wt + (size_t)n * K + (k0 + cg));
                float4 v0 = p[0], v1 = p[1];
                dst[0] = (_Float16)v0.x; dst[1] = (_Float16)v0.y;
                dst[2] = (_Float16)v0.z; dst[3] = (_Float16)v0.w;
                dst[4] = (_Float16)v1.x; dst[5] = (_Float16)v1.y;
                dst[6] = (_Float16)v1.z; dst[7] = (_Float16)v1.w;
            } else {
                #pragma unroll
                for (int j = 0; j < 8; ++j) dst[j] = (_Float16)0.0f;
            }
        }
        __syncthreads();

        const int half8 = (lane >> 4) * 8;
        const int r     = lane & 15;
        #pragma unroll
        for (int ks = 0; ks < 64; ks += 32) {
            h16v a0 = load_frag(&lA[(wm * 32 +      r) * 72 + ks], half8);
            h16v a1 = load_frag(&lA[(wm * 32 + 16 + r) * 72 + ks], half8);
            h16v b0 = load_frag(&lB[(wn * 32 +      r) * 72 + ks], half8);
            h16v b1 = load_frag(&lB[(wn * 32 + 16 + r) * 72 + ks], half8);
            acc00 = __builtin_amdgcn_wmma_f32_16x16x32_f16(false, a0, false, b0, (short)0, acc00, false, false);
            acc01 = __builtin_amdgcn_wmma_f32_16x16x32_f16(false, a0, false, b1, (short)0, acc01, false, false);
            acc10 = __builtin_amdgcn_wmma_f32_16x16x32_f16(false, a1, false, b0, (short)0, acc10, false, false);
            acc11 = __builtin_amdgcn_wmma_f32_16x16x32_f16(false, a1, false, b1, (short)0, acc11, false, false);
        }
        __syncthreads();
    }

    // Store: C/D layout: lane holds col n = lane&15; rows mb..mb+7 (lane>=16 -> +8)
    #pragma unroll
    for (int ti = 0; ti < 2; ++ti) {
        #pragma unroll
        for (int tj = 0; tj < 2; ++tj) {
            const f8v accv = (ti == 0) ? (tj == 0 ? acc00 : acc01)
                                       : (tj == 0 ? acc10 : acc11);
            const int n = n0 + wn * 32 + tj * 16 + (lane & 15);
            if (n < N) {
                const int   mb = m0 + wm * 32 + ti * 16 + (lane >> 4) * 8;
                const float bv = (bias != nullptr) ? bias[n] : 0.0f;
                #pragma unroll
                for (int rr = 0; rr < 8; ++rr) {
                    float v = accv[rr] + bv;
                    size_t idx = (size_t)(mb + rr) * ldC + n;
                    if (epi == 1)      v = 0.5f * v * (1.0f + erff(v * 0.70710678118654752f));
                    else if (epi == 2) v += resid[idx];
                    C[idx] = v;
                }
            }
        }
    }
}

// ---------------------------------------------------------------------------
// NCHW -> tokens + RMSNorm(norm1); writes x_tok, xn (h order), xn (v order)
// ---------------------------------------------------------------------------
__global__ __launch_bounds__(192)
void k_tok_norm1(const float* __restrict__ x, const float* __restrict__ n1w,
                 float* __restrict__ x_tok, float* __restrict__ xn_h,
                 float* __restrict__ xn_v)
{
    const int row = blockIdx.x;
    const int b = row / LSEQ, l = row % LSEQ;
    const int h = l / WW, w = l % WW;
    const int c = threadIdx.x;

    float v = x[(((size_t)b * DIMC + c) * HH + h) * WW + w];
    x_tok[(size_t)row * DIMC + c] = v;

    __shared__ float red[DIMC];
    __shared__ float invs;
    red[c] = v * v; __syncthreads();
    if (c < 96) red[c] += red[c + 96]; __syncthreads();
    if (c < 48) red[c] += red[c + 48]; __syncthreads();
    if (c < 24) red[c] += red[c + 24]; __syncthreads();
    if (c < 12) red[c] += red[c + 12]; __syncthreads();
    if (c <  6) red[c] += red[c +  6]; __syncthreads();
    if (c <  3) red[c] += red[c +  3]; __syncthreads();
    if (c == 0) invs = rsqrtf((red[0] + red[1] + red[2]) / (float)DIMC + 1e-6f);
    __syncthreads();

    float nv = n1w[c] * v * invs;
    xn_h[(size_t)row * DIMC + c] = nv;
    xn_v[((size_t)(b * LSEQ + w * HH + h)) * DIMC + c] = nv;
}

// ---------------------------------------------------------------------------
// Causal depthwise conv (k=4) + SiLU for both scan directions.
// ---------------------------------------------------------------------------
__global__ __launch_bounds__(256)
void k_conv_silu(const float* __restrict__ xz, const float* __restrict__ cw,
                 const float* __restrict__ cb,
                 float* __restrict__ u0, float* __restrict__ u1)
{
    const int i = blockIdx.x * blockDim.x + threadIdx.x;
    if (i >= ROWS_ * DINC) return;
    const int d = i % DINC;
    const int t = i / DINC;
    const int l = t % LSEQ;
    const int b = t / LSEQ;

    const float w0 = cw[d * 4 + 0], w1 = cw[d * 4 + 1];
    const float w2 = cw[d * 4 + 2], w3 = cw[d * 4 + 3];
    const float bias = cb[d];
    const float* xc = xz + (size_t)b * LSEQ * (2 * DINC) + d;

    float sf = bias;
    if (l >= 3) sf += w0 * xc[(size_t)(l - 3) * (2 * DINC)];
    if (l >= 2) sf += w1 * xc[(size_t)(l - 2) * (2 * DINC)];
    if (l >= 1) sf += w2 * xc[(size_t)(l - 1) * (2 * DINC)];
    sf += w3 * xc[(size_t)l * (2 * DINC)];
    u0[(size_t)t * DINC + d] = siluf(sf);

    float sb = bias;
    const int base = LSEQ - 1 - l;
    if (base + 3 < LSEQ) sb += w0 * xc[(size_t)(base + 3) * (2 * DINC)];
    if (base + 2 < LSEQ) sb += w1 * xc[(size_t)(base + 2) * (2 * DINC)];
    if (base + 1 < LSEQ) sb += w2 * xc[(size_t)(base + 1) * (2 * DINC)];
    sb += w3 * xc[(size_t)base * (2 * DINC)];
    u1[(size_t)t * DINC + d] = siluf(sb);
}

// ---------------------------------------------------------------------------
// Chunked selective scan, 3 passes. One lane per channel d.
// wid decode (passes 1/3): (((dir*2 + b)*NCHUNK + c)*12 + dchunk)
// Pass 1: per-chunk partial state (h from 0) and sum of deltas.
// Pass 2: serial combine over chunks -> per-chunk start states.
// Pass 3: re-run chunk from its start state, emit gated outputs.
// ---------------------------------------------------------------------------
__global__ __launch_bounds__(32)
void k_scan_p1(const float* __restrict__ xd0, const float* __restrict__ xd1,
               const float* __restrict__ u0,  const float* __restrict__ u1,
               const float* __restrict__ dtw, const float* __restrict__ dtb,
               const float* __restrict__ Alog,
               float* __restrict__ qbuf, float* __restrict__ sdel)
{
    const int wid    = blockIdx.x;              // 0..767
    const int dchunk = wid % 12;
    const int c      = (wid / 12) % NCHUNK;
    const int g      = wid / (12 * NCHUNK);     // dir*2 + b
    const int b      = g & 1;
    const int dir    = g >> 1;
    const int d      = dchunk * 32 + threadIdx.x;

    const float* xd = dir ? xd1 : xd0;
    const float* uu = dir ? u1  : u0;

    float wdt[DTRANK_];
    #pragma unroll
    for (int j = 0; j < DTRANK_; ++j) wdt[j] = dtw[d * DTRANK_ + j];
    float Ar[NSTATE_];
    #pragma unroll
    for (int n = 0; n < NSTATE_; ++n) Ar[n] = -__expf(Alog[d * NSTATE_ + n]);
    const float dtbd = dtb[d];

    float h[NSTATE_];
    #pragma unroll
    for (int n = 0; n < NSTATE_; ++n) h[n] = 0.0f;
    float sdelta = 0.0f;

    for (int l = c * CLEN; l < (c + 1) * CLEN; ++l) {
        const size_t row = (size_t)(b * LSEQ + l);
        const float* xr = xd + row * 48;
        float dt = dtbd;
        #pragma unroll
        for (int j = 0; j < DTRANK_; ++j) dt += xr[j] * wdt[j];
        const float delta = softplusf_(dt);
        sdelta += delta;
        const float du = delta * uu[row * DINC + d];
        #pragma unroll
        for (int n = 0; n < NSTATE_; ++n)
            h[n] = __expf(delta * Ar[n]) * h[n] + du * xr[12 + n];
    }

    const size_t qb = (((size_t)g * NCHUNK + c) * NSTATE_) * DINC;
    #pragma unroll
    for (int n = 0; n < NSTATE_; ++n) qbuf[qb + (size_t)n * DINC + d] = h[n];
    sdel[((size_t)g * NCHUNK + c) * DINC + d] = sdelta;
}

__global__ __launch_bounds__(32)
void k_scan_p2(const float* __restrict__ Alog, const float* __restrict__ qbuf,
               const float* __restrict__ sdel, float* __restrict__ hstart)
{
    const int wid    = blockIdx.x;      // 0..47 = g*12 + dchunk
    const int dchunk = wid % 12;
    const int g      = wid / 12;
    const int d      = dchunk * 32 + threadIdx.x;

    float Ar[NSTATE_];
    #pragma unroll
    for (int n = 0; n < NSTATE_; ++n) Ar[n] = -__expf(Alog[d * NSTATE_ + n]);

    float hs[NSTATE_];
    #pragma unroll
    for (int n = 0; n < NSTATE_; ++n) hs[n] = 0.0f;

    for (int c = 0; c < NCHUNK; ++c) {
        const size_t qb = (((size_t)g * NCHUNK + c) * NSTATE_) * DINC;
        const float sd = sdel[((size_t)g * NCHUNK + c) * DINC + d];
        #pragma unroll
        for (int n = 0; n < NSTATE_; ++n) {
            hstart[qb + (size_t)n * DINC + d] = hs[n];
            hs[n] = __expf(Ar[n] * sd) * hs[n] + qbuf[qb + (size_t)n * DINC + d];
        }
    }
}

__global__ __launch_bounds__(32)
void k_scan_p3(const float* __restrict__ xd0, const float* __restrict__ xd1,
               const float* __restrict__ u0,  const float* __restrict__ u1,
               const float* __restrict__ xz,
               const float* __restrict__ dtw, const float* __restrict__ dtb,
               const float* __restrict__ Alog, const float* __restrict__ Dp,
               const float* __restrict__ hstart,
               float* __restrict__ y0, float* __restrict__ y1)
{
    const int wid    = blockIdx.x;              // 0..767
    const int dchunk = wid % 12;
    const int c      = (wid / 12) % NCHUNK;
    const int g      = wid / (12 * NCHUNK);
    const int b      = g & 1;
    const int dir    = g >> 1;
    const int d      = dchunk * 32 + threadIdx.x;

    const float* xd = dir ? xd1 : xd0;
    const float* uu = dir ? u1  : u0;
    float*       yy = dir ? y1  : y0;

    float wdt[DTRANK_];
    #pragma unroll
    for (int j = 0; j < DTRANK_; ++j) wdt[j] = dtw[d * DTRANK_ + j];
    float Ar[NSTATE_];
    #pragma unroll
    for (int n = 0; n < NSTATE_; ++n) Ar[n] = -__expf(Alog[d * NSTATE_ + n]);
    const float dtbd = dtb[d];
    const float Dd   = Dp[d];

    const size_t qb = (((size_t)g * NCHUNK + c) * NSTATE_) * DINC;
    float h[NSTATE_];
    #pragma unroll
    for (int n = 0; n < NSTATE_; ++n) h[n] = hstart[qb + (size_t)n * DINC + d];

    for (int l = c * CLEN; l < (c + 1) * CLEN; ++l) {
        const size_t row = (size_t)(b * LSEQ + l);
        const float* xr = xd + row * 48;
        float dt = dtbd;
        #pragma unroll
        for (int j = 0; j < DTRANK_; ++j) dt += xr[j] * wdt[j];
        const float delta = softplusf_(dt);
        const float uv = uu[row * DINC + d];
        const float du = delta * uv;
        float y = 0.0f;
        #pragma unroll
        for (int n = 0; n < NSTATE_; ++n) {
            h[n] = __expf(delta * Ar[n]) * h[n] + du * xr[12 + n];
            y += h[n] * xr[28 + n];
        }
        const int zl = dir ? (LSEQ - 1 - l) : l;
        const float zv = xz[((size_t)(b * LSEQ + zl)) * (2 * DINC) + DINC + d];
        yy[row * DINC + d] = (y + uv * Dd) * siluf(zv);
    }
}

// y_comb[b,l,d] = y_fwd[b,l,d] + y_bwd[b,L-1-l,d]
__global__ __launch_bounds__(256)
void k_combine(const float* __restrict__ y0, const float* __restrict__ y1,
               float* __restrict__ yc)
{
    const int i = blockIdx.x * blockDim.x + threadIdx.x;
    if (i >= ROWS_ * DINC) return;
    const int d = i % DINC;
    const int t = i / DINC;
    const int l = t % LSEQ;
    const int b = t / LSEQ;
    yc[i] = y0[i] + y1[((size_t)(b * LSEQ + (LSEQ - 1 - l))) * DINC + d];
}

// ctx[b,c] = mean_l 0.5*(out_h + out_v)
__global__ __launch_bounds__(256)
void k_ctx(const float* __restrict__ oh, const float* __restrict__ ov,
           float* __restrict__ ctx)
{
    const int b = blockIdx.x / DIMC, c = blockIdx.x % DIMC;
    float s = 0.0f;
    for (int l = threadIdx.x; l < LSEQ; l += blockDim.x) {
        const size_t r = (size_t)(b * LSEQ + l) * DIMC + c;
        s += 0.5f * (oh[r] + ov[r]);
    }
    __shared__ float red[256];
    red[threadIdx.x] = s; __syncthreads();
    for (int st = 128; st > 0; st >>= 1) {
        if (threadIdx.x < st) red[threadIdx.x] += red[threadIdx.x + st];
        __syncthreads();
    }
    if (threadIdx.x == 0) ctx[blockIdx.x] = red[0] / (float)LSEQ;
}

// gate = sigmoid(relu(ctx @ gw1.T + gb1) @ gw2.T + gb2)
__global__ __launch_bounds__(256)
void k_gate(const float* __restrict__ ctx,
            const float* __restrict__ gw1, const float* __restrict__ gb1,
            const float* __restrict__ gw2, const float* __restrict__ gb2,
            float* __restrict__ gate)
{
    __shared__ float hid[48];
    for (int b = 0; b < BATCH_; ++b) {
        if (threadIdx.x < 48) {
            float s = gb1[threadIdx.x];
            for (int c = 0; c < DIMC; ++c) s += ctx[b * DIMC + c] * gw1[threadIdx.x * DIMC + c];
            hid[threadIdx.x] = fmaxf(s, 0.0f);
        }
        __syncthreads();
        if (threadIdx.x < DIMC) {
            float s = gb2[threadIdx.x];
            for (int j = 0; j < 48; ++j) s += hid[j] * gw2[threadIdx.x * 48 + j];
            gate[b * DIMC + threadIdx.x] = 1.0f / (1.0f + __expf(-s));
        }
        __syncthreads();
    }
}

// x_tok += gate*out_h + (1-gate)*out_v (in place); t_norm = RMSNorm(x_tok, norm2_w)
__global__ __launch_bounds__(192)
void k_fuse_norm2(float* __restrict__ x_tok, const float* __restrict__ oh,
                  const float* __restrict__ ov, const float* __restrict__ gate,
                  const float* __restrict__ n2w, float* __restrict__ t_norm)
{
    const int row = blockIdx.x;
    const int b = row / LSEQ;
    const int c = threadIdx.x;
    const size_t idx = (size_t)row * DIMC + c;

    const float g = gate[b * DIMC + c];
    float v = x_tok[idx] + g * oh[idx] + (1.0f - g) * ov[idx];
    x_tok[idx] = v;

    __shared__ float red[DIMC];
    __shared__ float invs;
    red[c] = v * v; __syncthreads();
    if (c < 96) red[c] += red[c + 96]; __syncthreads();
    if (c < 48) red[c] += red[c + 48]; __syncthreads();
    if (c < 24) red[c] += red[c + 24]; __syncthreads();
    if (c < 12) red[c] += red[c + 12]; __syncthreads();
    if (c <  6) red[c] += red[c +  6]; __syncthreads();
    if (c <  3) red[c] += red[c +  3]; __syncthreads();
    if (c == 0) invs = rsqrtf((red[0] + red[1] + red[2]) / (float)DIMC + 1e-6f);
    __syncthreads();

    t_norm[idx] = n2w[c] * v * invs;
}

// tokens [B,L,C] -> NCHW output
__global__ __launch_bounds__(256)
void k_to_nchw(const float* __restrict__ tok, float* __restrict__ out)
{
    const int i = blockIdx.x * blockDim.x + threadIdx.x;
    if (i >= BATCH_ * DIMC * HH * WW) return;
    const int w = i % WW;
    int t = i / WW;
    const int h = t % HH; t /= HH;
    const int c = t % DIMC;
    const int b = t / DIMC;
    out[i] = tok[((size_t)(b * LSEQ + h * WW + w)) * DIMC + c];
}

// ---------------------------------------------------------------------------
extern "C" void kernel_launch(void* const* d_in, const int* in_sizes, int n_in,
                              void* d_out, int out_size, void* d_ws, size_t ws_size,
                              hipStream_t stream)
{
    (void)in_sizes; (void)n_in; (void)out_size; (void)ws_size;

    const float* x   = (const float*)d_in[0];
    const float* n1w = (const float*)d_in[1];
    const float* n2w = (const float*)d_in[2];
    const float* gw1 = (const float*)d_in[21];
    const float* gb1 = (const float*)d_in[22];
    const float* gw2 = (const float*)d_in[23];
    const float* gb2 = (const float*)d_in[24];
    const float* mw1 = (const float*)d_in[25];
    const float* mb1 = (const float*)d_in[26];
    const float* mw2 = (const float*)d_in[27];
    const float* mb2 = (const float*)d_in[28];

    char* base = (char*)d_ws;
    size_t off = 0;
    auto take = [&](size_t bytes) -> float* {
        float* r = (float*)(base + off);
        off += (bytes + 255) & ~(size_t)255;
        return r;
    };
    float* x_tok = take((size_t)ROWS_ * DIMC * 4);
    float* xn_h  = take((size_t)ROWS_ * DIMC * 4);
    float* xn_v  = take((size_t)ROWS_ * DIMC * 4);
    float* out_h = take((size_t)ROWS_ * DIMC * 4);
    float* out_v = take((size_t)ROWS_ * DIMC * 4);
    float* ctx   = take(BATCH_ * DIMC * 4);
    float* gate  = take(BATCH_ * DIMC * 4);
    float* xz    = take((size_t)ROWS_ * 2 * DINC * 4);   // reused as MLP hidden
    float* u0    = take((size_t)ROWS_ * DINC * 4);       // reused as ycomb
    float* u1    = take((size_t)ROWS_ * DINC * 4);
    float* xd0   = take((size_t)ROWS_ * 48 * 4);
    float* xd1   = take((size_t)ROWS_ * 48 * 4);
    float* y0    = take((size_t)ROWS_ * DINC * 4);       // reused as t_norm
    float* y1    = take((size_t)ROWS_ * DINC * 4);       // reused as out_tok
    float* qbuf  = take((size_t)4 * NCHUNK * NSTATE_ * DINC * 4);
    float* hst   = take((size_t)4 * NCHUNK * NSTATE_ * DINC * 4);
    float* sdel  = take((size_t)4 * NCHUNK * DINC * 4);
    float* ycomb   = u0;
    float* mlp_hid = xz;
    float* t_norm  = y0;
    float* out_tok = y1;

    const dim3 blk(256);
    const int mtiles = ROWS_ / 128;  // 49

    k_tok_norm1<<<ROWS_, 192, 0, stream>>>(x, n1w, x_tok, xn_h, xn_v);

    for (int o = 0; o < 2; ++o) {
        const int pb = o ? 12 : 3;
        const float* in_w    = (const float*)d_in[pb + 0];
        const float* conv_w  = (const float*)d_in[pb + 1];
        const float* conv_b  = (const float*)d_in[pb + 2];
        const float* xproj_w = (const float*)d_in[pb + 3];
        const float* dt_w    = (const float*)d_in[pb + 4];
        const float* dt_b    = (const float*)d_in[pb + 5];
        const float* Alog    = (const float*)d_in[pb + 6];
        const float* Dp      = (const float*)d_in[pb + 7];
        const float* out_w   = (const float*)d_in[pb + 8];
        const float* xn      = o ? xn_v : xn_h;
        float*       out_o   = o ? out_v : out_h;

        // xz = xn @ in_w.T   [6272 x 768]
        k_gemm_wmma<<<dim3(768 / 64, mtiles), blk, 0, stream>>>(
            xn, in_w, xz, nullptr, nullptr, 2 * DINC, DIMC, 2 * DINC, 0);
        k_conv_silu<<<(ROWS_ * DINC + 255) / 256, blk, 0, stream>>>(xz, conv_w, conv_b, u0, u1);
        // x_dbl = u @ xproj_w.T   [6272 x 44], ld 48
        k_gemm_wmma<<<dim3(1, mtiles), blk, 0, stream>>>(
            u0, xproj_w, xd0, nullptr, nullptr, DTRANK_ + 2 * NSTATE_, DINC, 48, 0);
        k_gemm_wmma<<<dim3(1, mtiles), blk, 0, stream>>>(
            u1, xproj_w, xd1, nullptr, nullptr, DTRANK_ + 2 * NSTATE_, DINC, 48, 0);
        // chunked selective scan, 3 passes
        k_scan_p1<<<4 * NCHUNK * 12, 32, 0, stream>>>(xd0, xd1, u0, u1,
                                                      dt_w, dt_b, Alog, qbuf, sdel);
        k_scan_p2<<<48, 32, 0, stream>>>(Alog, qbuf, sdel, hst);
        k_scan_p3<<<4 * NCHUNK * 12, 32, 0, stream>>>(xd0, xd1, u0, u1, xz,
                                                      dt_w, dt_b, Alog, Dp, hst, y0, y1);
        k_combine<<<(ROWS_ * DINC + 255) / 256, blk, 0, stream>>>(y0, y1, ycomb);
        // out_o = ycomb @ out_w.T   [6272 x 192]
        k_gemm_wmma<<<dim3(192 / 64, mtiles), blk, 0, stream>>>(
            ycomb, out_w, out_o, nullptr, nullptr, DIMC, DINC, DIMC, 0);
    }

    k_ctx<<<BATCH_ * DIMC, blk, 0, stream>>>(out_h, out_v, ctx);
    k_gate<<<1, blk, 0, stream>>>(ctx, gw1, gb1, gw2, gb2, gate);
    k_fuse_norm2<<<ROWS_, 192, 0, stream>>>(x_tok, out_h, out_v, gate, n2w, t_norm);

    // MLP 192 -> 768 (GELU) -> 192 (+residual)
    k_gemm_wmma<<<dim3(768 / 64, mtiles), blk, 0, stream>>>(
        t_norm, mw1, mlp_hid, mb1, nullptr, 4 * DIMC, DIMC, 4 * DIMC, 1);
    k_gemm_wmma<<<dim3(192 / 64, mtiles), blk, 0, stream>>>(
        mlp_hid, mw2, out_tok, mb2, x_tok, DIMC, 4 * DIMC, DIMC, 2);

    k_to_nchw<<<(BATCH_ * DIMC * HH * WW + 255) / 256, blk, 0, stream>>>(
        out_tok, (float*)d_out);
}
